// VaswaniMultiHeadAttention_22565758174064
// MI455X (gfx1250) — matrix operand order repaired and checked by reading, not compiled, not run on hardware
//
#include <hip/hip_runtime.h>
#include <math.h>
#include <stdint.h>

// ---------------------------------------------------------------------------
// VaswaniMultiHeadAttention for MI455X (gfx1250), wave32 + WMMA f16->f32,
// with async global->LDS staging (ASYNCcnt) and double-buffered KV tiles.
// B=4, Sq=Skv=2048, E=1024, H=8, D=128.  scale = 1/sqrt(E) = 1/32.
// ---------------------------------------------------------------------------

typedef __attribute__((ext_vector_type(16))) _Float16 v16h;
typedef __attribute__((ext_vector_type(8)))  _Float16 v8h;
typedef __attribute__((ext_vector_type(8)))  float    v8f;

#define BATCH 4
#define SQ    2048
#define SKV   2048
#define EMB   1024
#define HEADS 8
#define HD    128

__device__ __forceinline__ v8f wmma_f16(v16h a, v16h b, v8f c) {
  // D = A(16x32 f16) * B(32x16 f16) + C(16x16 f32)
  return __builtin_amdgcn_wmma_f32_16x16x32_f16(
      /*neg_a=*/false, a, /*neg_b=*/false, b,
      /*c_mod=*/(short)0, c, /*reuse_a=*/false, /*reuse_b=*/false);
}

// Async DMA: 16 bytes global -> LDS, tracked with ASYNCcnt (gfx1250).
__device__ __forceinline__ void async_load_b128(uint32_t lds_off,
                                                const _Float16* gaddr) {
  asm volatile("global_load_async_to_lds_b128 %0, %1, off"
               :
               : "v"(lds_off), "v"(gaddr)
               : "memory");
}
__device__ __forceinline__ void wait_asynccnt0() {
  asm volatile("s_wait_asynccnt 0x0" ::: "memory");
}
__device__ __forceinline__ uint32_t lds_off(const void* p) {
  return (uint32_t)(uintptr_t)p;  // LDS aperture: low 32 bits = LDS address
}

// A-fragment (16x32, MxK) from f16 src, row-major [M][K], leading dim ld.
// ISA: lanes 0-15: M=lane, halves 0..7 = K k0..k0+7, halves 8..15 = k0+16..k0+23
//      lanes 16-31: M=lane-16, K offset +8 / +24.
__device__ __forceinline__ v16h load_a_frag(const _Float16* base, int ld,
                                            int m0, int k0) {
  const int lane = threadIdx.x & 31;
  const _Float16* p = base + (size_t)(m0 + (lane & 15)) * ld + k0 + ((lane >> 4) * 8);
  union { v16h v; v8h h[2]; } u;
  u.h[0] = *(const v8h*)(p);
  u.h[1] = *(const v8h*)(p + 16);
  return u.v;
}

// B-fragment (32x16, KxN) from "B-transposed" f16 storage [N][K] (K contiguous).
// ISA: lanes 0-15: N=lane, K = k0..k0+7, k0+16..k0+23; lanes 16-31: N=lane-16, K offset +8 / +24.
__device__ __forceinline__ v16h load_b_frag(const _Float16* baseT, int ld,
                                            int n0, int k0) {
  const int lane = threadIdx.x & 31;
  const _Float16* p = baseT + (size_t)(n0 + (lane & 15)) * ld + k0 + ((lane >> 4) * 8);
  union { v16h v; v8h h[2]; } u;
  u.h[0] = *(const v8h*)(p);
  u.h[1] = *(const v8h*)(p + 16);
  return u.v;
}

// A-fragment from f32 source (converted on the fly).
__device__ __forceinline__ v16h load_a_frag_f32(const float* base, int ld,
                                                int m0, int k0) {
  const int lane = threadIdx.x & 31;
  const float* p = base + (size_t)(m0 + (lane & 15)) * ld + k0 + ((lane >> 4) * 8);
  v16h r;
#pragma unroll
  for (int i = 0; i < 8; ++i) {
    r[i]     = (_Float16)p[i];
    r[i + 8] = (_Float16)p[i + 16];
  }
  return r;
}

// ---------------------------------------------------------------------------
// Projection GEMM:  out = f16( X[M,1024] @ W[1024,1024] + bias )
// transpose_out==0 : out row-major f16 [M][1024]
// transpose_out==1 : out f16 [B][1024][2048]   (per-batch transposed, for V')
// Block = 256 thr (8 waves). Block tile 128(M) x 32(N); wave tile 16 x 32.
// ---------------------------------------------------------------------------
__global__ __launch_bounds__(256) void proj_gemm_kernel(
    const float* __restrict__ X, const float* __restrict__ W,
    const float* __restrict__ bias, _Float16* __restrict__ out,
    int transpose_out) {
  __shared__ alignas(16) _Float16 Wt[32][40];  // W chunk transposed: [n][k]

  const int tid  = threadIdx.x;
  const int lane = tid & 31;
  const int wave = tid >> 5;
  const int n0   = blockIdx.x * 32;
  const int m0   = blockIdx.y * 128 + wave * 16;

  v8f acc0 = {}, acc1 = {};

  for (int kc = 0; kc < 1024; kc += 32) {
    // Stage W[kc..kc+31][n0..n0+31] transposed into LDS (each thread: 1 float4)
    {
      const int e  = tid * 4;       // 0..1023
      const int kk = e >> 5;        // 0..31
      const int nn = e & 31;        // multiple of 4
      const float4 wv = *(const float4*)(W + (size_t)(kc + kk) * 1024 + n0 + nn);
      Wt[nn + 0][kk] = (_Float16)wv.x;
      Wt[nn + 1][kk] = (_Float16)wv.y;
      Wt[nn + 2][kk] = (_Float16)wv.z;
      Wt[nn + 3][kk] = (_Float16)wv.w;
      if (kc + 32 < 1024)  // hint next W chunk into cache (global_prefetch)
        __builtin_prefetch(W + (size_t)(kc + 32 + kk) * 1024 + n0 + nn, 0, 1);
    }
    __syncthreads();

    const v16h a  = load_a_frag_f32(X, 1024, m0, kc);
    const v16h b0 = load_b_frag(&Wt[0][0], 40, 0, 0);
    const v16h b1 = load_b_frag(&Wt[0][0], 40, 16, 0);
    acc0 = wmma_f16(a, b0, acc0);
    acc1 = wmma_f16(a, b1, acc1);
    __syncthreads();
  }

  const int   ncol = lane & 15;
  const float bv0  = bias[n0 + ncol];
  const float bv1  = bias[n0 + 16 + ncol];
  const int   roff = (lane >> 4) * 8;
#pragma unroll
  for (int r = 0; r < 8; ++r) {
    const int   m  = m0 + roff + r;
    const float o0 = acc0[r] + bv0;
    const float o1 = acc1[r] + bv1;
    if (!transpose_out) {
      out[(size_t)m * 1024 + n0 + ncol]      = (_Float16)o0;
      out[(size_t)m * 1024 + n0 + 16 + ncol] = (_Float16)o1;
    } else {
      const int bb = m >> 11;        // batch (block tiles never straddle batches)
      const int ss = m & 2047;       // row within batch
      out[((size_t)bb * 1024 + n0 + ncol) * 2048 + ss]      = (_Float16)o0;
      out[((size_t)bb * 1024 + n0 + 16 + ncol) * 2048 + ss] = (_Float16)o1;
    }
  }
}

// ---------------------------------------------------------------------------
// Flash attention: block = (qblk, head, batch); 256 thr = 8 waves.
// Wave w: 16 query rows [qblk*128 + 16w, +16), full head dim 128.
// KV processed in chunks of 32 with online softmax; double-buffered LDS tiles
// filled by async global->LDS DMA overlapped with WMMA compute.
// ---------------------------------------------------------------------------
__global__ __launch_bounds__(256) void attn_kernel(
    const _Float16* __restrict__ QP,   // [B][SQ][EMB]  f16
    const _Float16* __restrict__ KP,   // [B][SKV][EMB] f16
    const _Float16* __restrict__ VPT,  // [B][EMB][SKV] f16 (transposed)
    float* __restrict__ out) {         // [B][SQ][EMB]  f32
  __shared__ alignas(16) _Float16 Klds[2][32][136];    // K chunk  [n][d]
  __shared__ alignas(16) _Float16 Vtlds[2][128][40];   // Vt chunk [d][n]
  __shared__ alignas(16) _Float16 Plds[8][16][40];     // per-wave P tile [m][n]

  const int tid  = threadIdx.x;
  const int lane = tid & 31;
  const int wave = tid >> 5;
  const int qblk = blockIdx.x;  // 0..15
  const int h    = blockIdx.y;  // 0..7
  const int b    = blockIdx.z;  // 0..3

  const int mbase = qblk * 128 + wave * 16;  // query row within batch

  // Per-thread staging coordinates (32B of K + 32B of Vt per thread per chunk)
  const int kr = tid >> 3;            // 0..31   (key row in chunk)
  const int kc = (tid & 7) * 16;      // 0..112  (d offset)
  const int vd = tid >> 1;            // 0..127  (d row)
  const int vc = (tid & 1) * 16;      // 0 / 16  (n offset)
  const _Float16* kbase =
      KP + ((size_t)b * SKV + kr) * EMB + (size_t)h * HD + kc;
  const _Float16* vbase =
      VPT + ((size_t)b * EMB + (size_t)h * HD + vd) * SKV + vc;

  // Q A-fragments: 4 chunks of K=32 covering head dim 128, kept in registers.
  v16h qa[4];
#pragma unroll
  for (int dc = 0; dc < 4; ++dc)
    qa[dc] = load_a_frag(QP + (size_t)b * SQ * EMB + (size_t)h * HD, EMB,
                         mbase, dc * 32);

  v8f acc[8];
  {
    v8f z = {};
#pragma unroll
    for (int i = 0; i < 8; ++i) acc[i] = z;
  }
  float mrow[8], lrow[8];
#pragma unroll
  for (int r = 0; r < 8; ++r) { mrow[r] = -1e30f; lrow[r] = 0.0f; }

  const float inv_scale = 0.03125f;            // 1/sqrt(1024)
  const float log2e     = 1.44269504088896f;

  // ---- prologue: async-stage chunk 0 into buffer 0 ----
  {
    async_load_b128(lds_off(&Klds[0][kr][kc]),     kbase);
    async_load_b128(lds_off(&Klds[0][kr][kc + 8]), kbase + 8);
    async_load_b128(lds_off(&Vtlds[0][vd][vc]),     vbase);
    async_load_b128(lds_off(&Vtlds[0][vd][vc + 8]), vbase + 8);
  }
  wait_asynccnt0();
  __syncthreads();

  int cur = 0;
  for (int n0 = 0; n0 < SKV; n0 += 32) {
    // ---- async prefetch of next chunk into the other buffer ----
    if (n0 + 32 < SKV) {
      const int nxt = cur ^ 1;
      const _Float16* kp = kbase + (size_t)(n0 + 32) * EMB;
      const _Float16* vp = vbase + (n0 + 32);
      async_load_b128(lds_off(&Klds[nxt][kr][kc]),     kp);
      async_load_b128(lds_off(&Klds[nxt][kr][kc + 8]), kp + 8);
      async_load_b128(lds_off(&Vtlds[nxt][vd][vc]),     vp);
      async_load_b128(lds_off(&Vtlds[nxt][vd][vc + 8]), vp + 8);
    }

    // ---- scores: S(16x32) = Q(16x128) * K^T, two N-frags, 4 K-chunks ----
    v8f s0 = {}, s1 = {};
#pragma unroll
    for (int dc = 0; dc < 4; ++dc) {
      const v16h kb0 = load_b_frag(&Klds[cur][0][0], 136, 0, dc * 32);
      const v16h kb1 = load_b_frag(&Klds[cur][0][0], 136, 16, dc * 32);
      s0 = wmma_f16(qa[dc], kb0, s0);
      s1 = wmma_f16(qa[dc], kb1, s1);
    }

    // ---- online softmax (row stats per VGPR r, per 16-lane half) ----
    float scalef[8];
#pragma unroll
    for (int r = 0; r < 8; ++r) {
      const float s0r = s0[r] * inv_scale;
      const float s1r = s1[r] * inv_scale;
      float mx = fmaxf(s0r, s1r);
      mx = fmaxf(mx, __shfl_xor(mx, 1, 32));
      mx = fmaxf(mx, __shfl_xor(mx, 2, 32));
      mx = fmaxf(mx, __shfl_xor(mx, 4, 32));
      mx = fmaxf(mx, __shfl_xor(mx, 8, 32));
      const float mn = fmaxf(mrow[r], mx);
      const float p0 = exp2f((s0r - mn) * log2e);
      const float p1 = exp2f((s1r - mn) * log2e);
      float ps = p0 + p1;
      ps += __shfl_xor(ps, 1, 32);
      ps += __shfl_xor(ps, 2, 32);
      ps += __shfl_xor(ps, 4, 32);
      ps += __shfl_xor(ps, 8, 32);
      const float corr = exp2f((mrow[r] - mn) * log2e);
      lrow[r]   = lrow[r] * corr + ps;
      mrow[r]   = mn;
      scalef[r] = corr;
      // stash P in per-wave LDS tile, [m][n] row-major (feeds A-fragments)
      const int m  = r + ((lane >> 4) * 8);
      const int nn = lane & 15;
      Plds[wave][m][nn]      = (_Float16)p0;
      Plds[wave][m][16 + nn] = (_Float16)p1;
    }

    // rescale accumulators by exp(m_old - m_new) per row
#pragma unroll
    for (int df = 0; df < 8; ++df)
#pragma unroll
      for (int r = 0; r < 8; ++r) acc[df][r] *= scalef[r];

    // ---- AV: out(16x128) += P(16x32) * V(32x128); one shared A-frag ----
    const v16h pa = load_a_frag(&Plds[wave][0][0], 40, 0, 0);
#pragma unroll
    for (int df = 0; df < 8; ++df) {
      const v16h vb = load_b_frag(&Vtlds[cur][0][0], 40, df * 16, 0);
      acc[df] = wmma_f16(pa, vb, acc[df]);
    }

    // my async DMA into the other buffer must be complete before the barrier;
    // the barrier then both publishes it and retires everyone's reads of `cur`.
    wait_asynccnt0();
    __syncthreads();
    cur ^= 1;
  }

  // ---- finalize: divide by l, store f32 ----
  const int ncol = lane & 15;
  const int roff = (lane >> 4) * 8;
#pragma unroll
  for (int r = 0; r < 8; ++r) {
    const float invl = 1.0f / lrow[r];
    const int   m    = mbase + roff + r;
    const size_t base = ((size_t)b * SQ + m) * EMB + (size_t)h * HD;
#pragma unroll
    for (int df = 0; df < 8; ++df)
      out[base + df * 16 + ncol] = acc[df][r] * invl;
  }
}

// ---------------------------------------------------------------------------
extern "C" void kernel_launch(void* const* d_in, const int* in_sizes, int n_in,
                              void* d_out, int out_size, void* d_ws,
                              size_t ws_size, hipStream_t stream) {
  const float* q  = (const float*)d_in[0];  // [4,2048,1024]
  const float* v  = (const float*)d_in[1];  // [4,2048,1024]
  const float* Wq = (const float*)d_in[2];
  const float* bq = (const float*)d_in[3];
  const float* Wk = (const float*)d_in[4];
  const float* bk = (const float*)d_in[5];
  const float* Wv = (const float*)d_in[6];
  const float* bv = (const float*)d_in[7];
  float* out = (float*)d_out;

  const size_t elems = (size_t)BATCH * SQ * EMB;  // 8M elements
  _Float16* QP  = (_Float16*)d_ws;                // 16 MB
  _Float16* KP  = QP + elems;                     // 16 MB
  _Float16* VPT = KP + elems;                     // 16 MB (transposed V')
  (void)in_sizes; (void)n_in; (void)out_size; (void)ws_size;

  dim3 pg(1024 / 32, (BATCH * SQ) / 128);  // 32 x 64 blocks
  proj_gemm_kernel<<<pg, 256, 0, stream>>>(q, Wq, bq, QP, 0);
  proj_gemm_kernel<<<pg, 256, 0, stream>>>(v, Wk, bk, KP, 0);
  proj_gemm_kernel<<<pg, 256, 0, stream>>>(v, Wv, bv, VPT, 1);

  dim3 ag(SQ / 128, HEADS, BATCH);  // 16 x 8 x 4 blocks
  attn_kernel<<<ag, 256, 0, stream>>>(QP, KP, VPT, out);
}
